// DeepESNGatedGRU_62912680952224
// MI455X (gfx1250) — matrix operand order repaired and checked by reading, not compiled, not run on hardware
//
#include <hip/hip_runtime.h>
#include <cstdint>
#include <cstddef>

typedef __bf16 bf16_t;
typedef __attribute__((ext_vector_type(16))) __bf16 v16bf;
typedef __attribute__((ext_vector_type(8)))  float  v8f;
typedef int v4i_ __attribute__((vector_size(16)));

#define NBLK   16
#define TPB    256
#define BATCH  64
#define HDIM   512
#define TSTEPS 512
#define INDIM  256
#define OUTDIM 10

#if __has_builtin(__builtin_amdgcn_global_load_async_to_lds_b128)
#define ASYNC_LDS 1
#endif

// ---------------------------------------------------------------------------
// WMMA fragment loaders (bf16, 16x16x32, wave32)
// A (16x32, MxK): lane<16 -> row M=lane,    K chunks [k0+0..7],  [k0+16..23]
//                 lane>=16-> row M=lane-16, K chunks [k0+8..15], [k0+24..31]
// B (32x16, KxN): lane<16 -> col N=lane,    K chunks [k0+0..7],  [k0+16..23]
//                 lane>=16-> col N=lane-16, K chunks [k0+8..15], [k0+24..31]
// Weights stored transposed (N x K row-major bf16): B = two b128 loads/lane.
// A operands are staged in LDS, so A fragment loads become ds_load_b128.
// ---------------------------------------------------------------------------
__device__ __forceinline__ v16bf load_a(const bf16_t* A, int ldA, int m_base,
                                        int k0, int lane) {
  const int half = lane >> 4;
  const bf16_t* p = A + (size_t)(m_base + (lane & 15)) * ldA + k0 + half * 8;
  union { v16bf v; uint4 q[2]; } u;
  u.q[0] = *(const uint4*)(p);
  u.q[1] = *(const uint4*)(p + 16);
  return u.v;
}

__device__ __forceinline__ v16bf load_b(const bf16_t* WT, int Ktot, int n_base,
                                        int k0, int lane) {
  const int half = lane >> 4;
  const bf16_t* p = WT + (size_t)(n_base + (lane & 15)) * Ktot + k0 + half * 8;
  union { v16bf v; uint4 q[2]; } u;
  u.q[0] = *(const uint4*)(p);
  u.q[1] = *(const uint4*)(p + 16);
  return u.v;
}

// acc += A_seg(16xKseg) @ W_seg(Ksegx16) for this wave's tile
__device__ __forceinline__ void gemm_seg(v8f& acc, const bf16_t* A, int ldA,
                                         const bf16_t* WT, int Ktot, int koff,
                                         int Kseg, int m_base, int n_base,
                                         int lane) {
  for (int k0 = 0; k0 < Kseg; k0 += 32) {
    v16bf a = load_a(A, ldA, m_base, k0, lane);
    v16bf b = load_b(WT, Ktot, n_base, koff + k0, lane);
    acc = __builtin_amdgcn_wmma_f32_16x16x32_bf16(false, a, false, b,
                                                  (short)0, acc, false, false);
  }
}

// Two gates sharing the same A operand (z and r): one A load, two WMMAs
__device__ __forceinline__ void gemm_seg2(v8f& acc0, v8f& acc1,
                                          const bf16_t* A, int ldA,
                                          const bf16_t* WT0, const bf16_t* WT1,
                                          int Ktot, int koff, int Kseg,
                                          int m_base, int n_base, int lane) {
  for (int k0 = 0; k0 < Kseg; k0 += 32) {
    v16bf a  = load_a(A, ldA, m_base, k0, lane);
    v16bf b0 = load_b(WT0, Ktot, n_base, koff + k0, lane);
    v16bf b1 = load_b(WT1, Ktot, n_base, koff + k0, lane);
    acc0 = __builtin_amdgcn_wmma_f32_16x16x32_bf16(false, a, false, b0,
                                                   (short)0, acc0, false, false);
    acc1 = __builtin_amdgcn_wmma_f32_16x16x32_bf16(false, a, false, b1,
                                                   (short)0, acc1, false, false);
  }
}

// ---------------------------------------------------------------------------
// Stage a 64xK bf16 matrix (global) into LDS. Async path uses the CDNA5
// GLOBAL_LOAD_ASYNC_TO_LDS_B128 path (ASYNCcnt); fallback is b128 load +
// ds_store_b128. elems must be a multiple of 8.
// ---------------------------------------------------------------------------
__device__ __forceinline__ void stage_to_lds(bf16_t* dst_lds,
                                             const bf16_t* src_g, int elems) {
  const int chunks = elems >> 3;  // 16 bytes each
  for (int c = threadIdx.x; c < chunks; c += TPB) {
#ifdef ASYNC_LDS
    __builtin_amdgcn_global_load_async_to_lds_b128(
        (v4i_*)(const_cast<bf16_t*>(src_g) + c * 8),
        (v4i_*)(dst_lds + c * 8), 0, 0);
#else
    *(uint4*)(dst_lds + c * 8) = *(const uint4*)(src_g + c * 8);
#endif
  }
}

__device__ __forceinline__ void wait_stage() {
#ifdef ASYNC_LDS
#if __has_builtin(__builtin_amdgcn_s_wait_asynccnt)
  __builtin_amdgcn_s_wait_asynccnt(0);
#else
  asm volatile("s_wait_asynccnt 0x0" ::: "memory");
#endif
#endif
  __syncthreads();
}

// ---------------------------------------------------------------------------
// Cross-workgroup sync: global-atomic monotonic barrier (correct under plain
// launch) + CDNA5 cluster split-barrier fast path (NOP when not clustered).
// ---------------------------------------------------------------------------
__device__ __forceinline__ void grid_sync(unsigned* cnt, unsigned phase) {
  __syncthreads();
  if (threadIdx.x == 0) {
    __threadfence();
    atomicAdd(cnt, 1u);
    const unsigned target = phase * (unsigned)NBLK;
    while (atomicAdd(cnt, 0u) < target) __builtin_amdgcn_s_sleep(2);
  }
  if (threadIdx.x < 32) {
    asm volatile("s_barrier_signal -3" ::: "memory");
  }
  asm volatile("s_barrier_wait -3" ::: "memory");
  __syncthreads();
  __threadfence();
}

__device__ __forceinline__ float sigmoidf_(float x) {
  return 1.0f / (1.0f + __expf(-x));
}

// ---------------------------------------------------------------------------
// Persistent per-layer recurrence kernel: 16 WGs x 256 threads (8 wave32)
// wave w: m_tile = w>>1 (0..3), n_tile local = w&1; block owns 32 N columns.
// ---------------------------------------------------------------------------
struct LayerParams {
  const bf16_t* xseq;   // [T][64][lin] bf16
  int lin;
  const bf16_t *WinT, *WresT, *WzT, *WrT, *WcT, *WgT, *WpT;  // [512][K] bf16
  const float *bz, *br, *bc, *bg, *bp;                        // [512] f32
  float *h, *r0, *r1, *zbuf, *ptbuf, *hgruf;                  // [64*512] f32
  bf16_t *hb, *r0b, *r1b, *rhb, *hgrub;                       // [64*512] bf16
  bf16_t *outseq;       // [T][64][512] bf16 (nullptr for last layer)
  unsigned* sync;
};

__global__ __launch_bounds__(TPB) void esn_layer_kernel(LayerParams P) {
  // 192KB static LDS: x_t persists for the whole step; s0/s1 are per-phase
  // staging buffers for the recurrent-state A operands.
  __shared__ __align__(16) bf16_t lds_x[BATCH * HDIM];   // 64KB (64 x lin used)
  __shared__ __align__(16) bf16_t lds_s0[BATCH * HDIM];  // 64KB
  __shared__ __align__(16) bf16_t lds_s1[BATCH * HDIM];  // 64KB

  const int tid  = threadIdx.x;
  const int lane = tid & 31;
  const int wid  = tid >> 5;
  const int m_base = (wid >> 1) << 4;
  const int n_base = blockIdx.x * 32 + ((wid & 1) << 4);
  const int lin = P.lin;
  unsigned phase = 1;

  // zero recurrent state
  for (int i = blockIdx.x * TPB + tid; i < BATCH * HDIM; i += NBLK * TPB) {
    P.h[i] = 0.f;  P.r0[i] = 0.f;  P.r1[i] = 0.f;
    P.hb[i]  = (bf16_t)0.f;
    P.r0b[i] = (bf16_t)0.f;
    P.r1b[i] = (bf16_t)0.f;
  }
  grid_sync(P.sync, phase++);

  for (int t = 0; t < TSTEPS; ++t) {
    const bf16_t* xt = P.xseq + (size_t)t * BATCH * lin;
    float*  rprev  = (t & 1) ? P.r1  : P.r0;
    float*  rnext  = (t & 1) ? P.r0  : P.r1;
    bf16_t* rprevb = (t & 1) ? P.r1b : P.r0b;
    bf16_t* rnextb = (t & 1) ? P.r0b : P.r1b;

    if (t + 1 < TSTEPS)   // pull next step's activations toward L2
      __builtin_prefetch(P.xseq + (size_t)(t + 1) * BATCH * lin + tid * 32, 0, 3);

    // ---- Phase 1: reservoir pre-activation, z gate, reset gate ------------
    stage_to_lds(lds_x, xt, BATCH * lin);          // persists across phases
    stage_to_lds(lds_s0, P.hb, BATCH * HDIM);      // h_{t-1}
    stage_to_lds(lds_s1, rprevb, BATCH * HDIM);    // r_{t-1}
    wait_stage();

    v8f accRes = {0,0,0,0,0,0,0,0};
    v8f accZ   = {0,0,0,0,0,0,0,0};
    v8f accR   = {0,0,0,0,0,0,0,0};
    gemm_seg (accRes, lds_x,  lin,  P.WinT,  lin,        0,   lin,  m_base, n_base, lane);
    gemm_seg (accRes, lds_s1, HDIM, P.WresT, HDIM,       0,   HDIM, m_base, n_base, lane);
    gemm_seg2(accZ, accR, lds_x,  lin,  P.WzT, P.WrT, lin+HDIM, 0,   lin,  m_base, n_base, lane);
    gemm_seg2(accZ, accR, lds_s0, HDIM, P.WzT, P.WrT, lin+HDIM, lin, HDIM, m_base, n_base, lane);

#pragma unroll
    for (int j = 0; j < 8; ++j) {
      const int m = m_base + j + ((lane >> 4) << 3);
      const int n = n_base + (lane & 15);
      const size_t idx = (size_t)m * HDIM + n;
      const float rn = 0.7f * rprev[idx] + 0.3f * tanhf(accRes[j]);
      rnext[idx]  = rn;
      rnextb[idx] = (bf16_t)rn;
      const float z = sigmoidf_(accZ[j] + P.bz[n]);
      P.zbuf[idx] = z;
      const float rst = sigmoidf_(accR[j] + P.br[n]);
      P.rhb[idx] = (bf16_t)(rst * P.h[idx]);
    }
    grid_sync(P.sync, phase++);

    // ---- Phase 2: candidate, h_gru, projection tanh(r@Wp+bp) --------------
    stage_to_lds(lds_s0, P.rhb, BATCH * HDIM);     // reset * h
    stage_to_lds(lds_s1, rnextb, BATCH * HDIM);    // r_t (kept for phase 3)
    wait_stage();

    v8f accC = {0,0,0,0,0,0,0,0};
    v8f accP = {0,0,0,0,0,0,0,0};
    gemm_seg(accC, lds_x,  lin,  P.WcT, lin+HDIM, 0,   lin,  m_base, n_base, lane);
    gemm_seg(accC, lds_s0, HDIM, P.WcT, lin+HDIM, lin, HDIM, m_base, n_base, lane);
    gemm_seg(accP, lds_s1, HDIM, P.WpT, HDIM,     0,   HDIM, m_base, n_base, lane);

#pragma unroll
    for (int j = 0; j < 8; ++j) {
      const int m = m_base + j + ((lane >> 4) << 3);
      const int n = n_base + (lane & 15);
      const size_t idx = (size_t)m * HDIM + n;
      const float c  = tanhf(accC[j] + P.bc[n]);
      const float z  = P.zbuf[idx];
      const float hg = (1.0f - z) * P.h[idx] + z * c;
      P.hgruf[idx] = hg;
      P.hgrub[idx] = (bf16_t)hg;
      P.ptbuf[idx] = tanhf(accP[j] + P.bp[n]);
    }
    grid_sync(P.sync, phase++);

    // ---- Phase 3: g gate over [x, h_gru, r_new], final h blend ------------
    stage_to_lds(lds_s0, P.hgrub, BATCH * HDIM);   // h_gru (lds_s1 still = r_t)
    wait_stage();

    v8f accG = {0,0,0,0,0,0,0,0};
    gemm_seg(accG, lds_x,  lin,  P.WgT, lin+2*HDIM, 0,        lin,  m_base, n_base, lane);
    gemm_seg(accG, lds_s0, HDIM, P.WgT, lin+2*HDIM, lin,      HDIM, m_base, n_base, lane);
    gemm_seg(accG, lds_s1, HDIM, P.WgT, lin+2*HDIM, lin+HDIM, HDIM, m_base, n_base, lane);

#pragma unroll
    for (int j = 0; j < 8; ++j) {
      const int m = m_base + j + ((lane >> 4) << 3);
      const int n = n_base + (lane & 15);
      const size_t idx = (size_t)m * HDIM + n;
      const float g  = sigmoidf_(accG[j] + P.bg[n]);
      const float hn = (1.0f - g) * P.hgruf[idx] + g * P.ptbuf[idx];
      P.h[idx]  = hn;
      P.hb[idx] = (bf16_t)hn;
      if (P.outseq) P.outseq[(size_t)t * BATCH * HDIM + idx] = (bf16_t)hn;
    }
    grid_sync(P.sync, phase++);
  }
}

// ---------------------------------------------------------------------------
// Support kernels
// ---------------------------------------------------------------------------
__global__ void zero_u32_kernel(unsigned* p, int n) {
  int i = blockIdx.x * blockDim.x + threadIdx.x;
  if (i < n) p[i] = 0u;
}

// x[b][t][i] (fp32, B,T,IN) -> xseq[t][b][i] (bf16)
__global__ void xcvt_kernel(const float* __restrict__ x, bf16_t* __restrict__ o,
                            int total) {
  int idx = blockIdx.x * blockDim.x + threadIdx.x;
  if (idx >= total) return;
  const int i = idx % INDIM;
  const int b = (idx / INDIM) % BATCH;
  const int t = idx / (INDIM * BATCH);
  o[idx] = (bf16_t)x[((size_t)b * TSTEPS + t) * INDIM + i];
}

// W (KxN fp32 row-major) -> WT (NxK bf16 row-major)
__global__ void wt_kernel(const float* __restrict__ W, bf16_t* __restrict__ WT,
                          int K, int N, int total) {
  int idx = blockIdx.x * blockDim.x + threadIdx.x;
  if (idx >= total) return;
  const int n = idx / K;
  const int k = idx % K;
  WT[idx] = (bf16_t)W[(size_t)k * N + n];
}

__global__ void head1_kernel(const float* __restrict__ h,
                             const float* __restrict__ Wo1,
                             const float* __restrict__ bo1,
                             float* __restrict__ tmp) {
  int idx = blockIdx.x * blockDim.x + threadIdx.x;
  if (idx >= BATCH * HDIM) return;
  const int b = idx >> 9, n = idx & (HDIM - 1);
  float s = bo1[n];
  for (int k = 0; k < HDIM; ++k) s += h[(size_t)b * HDIM + k] * Wo1[(size_t)k * HDIM + n];
  tmp[idx] = fmaxf(s, 0.0f);
}

__global__ void head2_kernel(const float* __restrict__ tmp,
                             const float* __restrict__ Wo2,
                             const float* __restrict__ bo2,
                             float* __restrict__ out) {
  int idx = blockIdx.x * blockDim.x + threadIdx.x;
  if (idx >= BATCH * OUTDIM) return;
  const int b = idx / OUTDIM, o = idx % OUTDIM;
  float s = bo2[o];
  for (int k = 0; k < HDIM; ++k) s += tmp[(size_t)b * HDIM + k] * Wo2[(size_t)k * OUTDIM + o];
  out[idx] = s;
}

// ---------------------------------------------------------------------------
extern "C" void kernel_launch(void* const* d_in, const int* in_sizes, int n_in,
                              void* d_out, int out_size, void* d_ws, size_t ws_size,
                              hipStream_t stream) {
  (void)in_sizes; (void)n_in; (void)out_size; (void)ws_size;
  const float* W[29];
  for (int i = 0; i < 29; ++i) W[i] = (const float*)d_in[i];

  char* ws = (char*)d_ws;
  size_t off = 0;
  auto alloc = [&](size_t bytes) -> void* {
    off = (off + 255) & ~(size_t)255;
    void* p = ws + off;
    off += bytes;
    return p;
  };

  bf16_t* xseq0 = (bf16_t*)alloc((size_t)TSTEPS * BATCH * INDIM * sizeof(bf16_t));
  bf16_t* seq1  = (bf16_t*)alloc((size_t)TSTEPS * BATCH * HDIM  * sizeof(bf16_t));

  // Win, Wres, Wz, Wr, Wc, Wg, Wp per layer (d_in indices; K per weight)
  const int widx[14] = {1, 2, 3, 5, 7, 9, 11, 13, 14, 15, 17, 19, 21, 23};
  const int wK[14]   = {256, 512, 768, 768, 768, 1792, 512,
                        512, 512, 1024, 1024, 1024, 2048, 512};
  bf16_t* wt[14];
  for (int i = 0; i < 14; ++i)
    wt[i] = (bf16_t*)alloc((size_t)wK[i] * HDIM * sizeof(bf16_t));

  float*  hstate = (float*) alloc(BATCH * HDIM * 4);
  bf16_t* hb     = (bf16_t*)alloc(BATCH * HDIM * 2);
  float*  r0     = (float*) alloc(BATCH * HDIM * 4);
  float*  r1     = (float*) alloc(BATCH * HDIM * 4);
  bf16_t* r0b    = (bf16_t*)alloc(BATCH * HDIM * 2);
  bf16_t* r1b    = (bf16_t*)alloc(BATCH * HDIM * 2);
  float*  zbuf   = (float*) alloc(BATCH * HDIM * 4);
  float*  ptbuf  = (float*) alloc(BATCH * HDIM * 4);
  bf16_t* rhb    = (bf16_t*)alloc(BATCH * HDIM * 2);
  float*  hgruf  = (float*) alloc(BATCH * HDIM * 4);
  bf16_t* hgrub  = (bf16_t*)alloc(BATCH * HDIM * 2);
  float*  headtmp= (float*) alloc(BATCH * HDIM * 4);
  unsigned* syncc= (unsigned*)alloc(2 * sizeof(unsigned));

  zero_u32_kernel<<<1, 32, 0, stream>>>(syncc, 2);
  {
    const int total = BATCH * TSTEPS * INDIM;
    xcvt_kernel<<<(total + TPB - 1) / TPB, TPB, 0, stream>>>(W[0], xseq0, total);
  }
  for (int i = 0; i < 14; ++i) {
    const int total = wK[i] * HDIM;
    wt_kernel<<<(total + TPB - 1) / TPB, TPB, 0, stream>>>(W[widx[i]], wt[i],
                                                           wK[i], HDIM, total);
  }

  LayerParams P0;
  P0.xseq = xseq0;  P0.lin = INDIM;
  P0.WinT = wt[0];  P0.WresT = wt[1]; P0.WzT = wt[2]; P0.WrT = wt[3];
  P0.WcT  = wt[4];  P0.WgT  = wt[5];  P0.WpT = wt[6];
  P0.bz = W[4];  P0.br = W[6];  P0.bc = W[8];  P0.bg = W[10]; P0.bp = W[12];
  P0.h = hstate; P0.r0 = r0; P0.r1 = r1; P0.zbuf = zbuf; P0.ptbuf = ptbuf;
  P0.hgruf = hgruf; P0.hb = hb; P0.r0b = r0b; P0.r1b = r1b; P0.rhb = rhb;
  P0.hgrub = hgrub; P0.outseq = seq1; P0.sync = syncc + 0;
  esn_layer_kernel<<<NBLK, TPB, 0, stream>>>(P0);

  LayerParams P1 = P0;
  P1.xseq = seq1;  P1.lin = HDIM;
  P1.WinT = wt[7]; P1.WresT = wt[8]; P1.WzT = wt[9]; P1.WrT = wt[10];
  P1.WcT  = wt[11]; P1.WgT = wt[12]; P1.WpT = wt[13];
  P1.bz = W[16]; P1.br = W[18]; P1.bc = W[20]; P1.bg = W[22]; P1.bp = W[24];
  P1.outseq = nullptr; P1.sync = syncc + 1;
  esn_layer_kernel<<<NBLK, TPB, 0, stream>>>(P1);

  head1_kernel<<<(BATCH * HDIM + TPB - 1) / TPB, TPB, 0, stream>>>(hstate, W[25], W[26], headtmp);
  head2_kernel<<<(BATCH * OUTDIM + TPB - 1) / TPB, TPB, 0, stream>>>(headtmp, W[27], W[28], (float*)d_out);
}
